// GCN_19232863552045
// MI455X (gfx1250) — matrix operand order repaired and checked by reading, not compiled, not run on hardware
//
#include <hip/hip_runtime.h>
#include <hip/hip_bf16.h>

// ---------------------------------------------------------------------------
// Problem constants (match reference setup_inputs()).
// ---------------------------------------------------------------------------
#define NN     10000      // nodes
#define PP     2000       // proteins
#define CT     10         // cell types
#define EE     320000     // edges
#define DAE0   100
#define DAE1   20
#define GAE0   32
#define GAE1   20
#define FCAT   40         // DAE1 + GAE1
#define FEAT   64
#define MAXLS  10.0f
#define NEPS   1e-6f

// output layout (floats): ct | x_dec | gae_loss | pw
#define CT_OFF    0
#define XDEC_OFF  (NN * CT)                        // 100000
#define GAE_OFF   (XDEC_OFF + (size_t)NN * PP)     // 20,100,000
#define PW_OFF    (GAE_OFF + 1)                    // 20,100,001

typedef __attribute__((ext_vector_type(16))) _Float16 v16h;
typedef __attribute__((ext_vector_type(8)))  _Float16 v8h;
typedef __attribute__((ext_vector_type(2)))  _Float16 v2h;
typedef __attribute__((ext_vector_type(8)))  float    v8f;

#define ACT_NONE 0
#define ACT_CELU 1
#define ACT_RELU 2

// K-run stride in LDS: 32 data halfs + 8 pad (80B) -> bank-conflict-free b128
// fragment reads (each of 64 banks hit exactly twice per wave instruction)
// while preserving 16B alignment.
#define CST 40

// ---------------------------------------------------------------------------
// Wave-tile GEMM: D[M,N] = act(A[M,K] @ B[K,N] + bias[N]), f16 WMMA, f32 acc.
//
// Block = 256 threads (8 waves). The block's B n-strip is staged into LDS
// once, PRE-TRANSPOSED into the WMMA B-fragment layout:
//     Bs[kchunk][col][krun 0..31 (+8 pad)]   (f16, K zero-padded to KP)
// so each lane's 16 B halfs are contiguous -> fragment = 2x ds_load_b128.
// A fragment = four float4 (b128) global loads + packed f32->f16 converts.
// All loads branchless (clamped addresses; garbage only reaches guarded
// stores); only the A fragment needs K-tail value-selects (B pad is zero).
// Requires K % 4 == 0 (call sites: 2000, 100, 20, 32, 40).
// Lane layouts used here (contiguous k, shared by A and B; not the ISA table order):
//   A 16x32 f16 : h = lane>>4, row = lane&15, half e -> K = h*16 + e
//   B 32x16 f16 : col = lane&15, half e -> K = h*16 + e
//   C/D 16x16 f32: vgpr r -> row = r + 8*h, col = lane&15
// grid = ( ceil(N/(16*NT)), ceil(M/128) ), dyn LDS = (KP/32)*16*NT*CST*2 B.
// ---------------------------------------------------------------------------
template <int NT>
__global__ __launch_bounds__(256)
void k_gemm_wmma(const float* __restrict__ A,
                 const float* __restrict__ B,
                 const float* __restrict__ bias,
                 float* __restrict__ D,
                 int M, int N, int K, int act)
{
    extern __shared__ _Float16 Bs[];
    constexpr int W = 16 * NT;

    const int tid  = threadIdx.x;
    const int wave = tid >> 5;
    const int lane = tid & 31;
    const int half = lane >> 4;
    const int lrow = lane & 15;

    const int n0 = blockIdx.x * W;
    const int m0 = (blockIdx.y * 8 + wave) * 16;
    const int KP = (K + 31) & ~31;

    // ---- cooperative stage: B[n-strip] -> LDS fragment layout, zero-pad K --
    for (int idx = tid; idx < (KP >> 1) * W; idx += 256) {
        const int kh = idx / W;              // K-pair index
        const int j  = idx - kh * W;
        const int k2 = kh * 2;
        int c = n0 + j; c = (c < N) ? c : (N - 1);
        const float f0 = B[(size_t)((k2 < K) ? k2 : (K - 1)) * N + c];
        const float f1 = B[(size_t)((k2 + 1 < K) ? (k2 + 1) : (K - 1)) * N + c];
        v2h pr;
        pr.x = (_Float16)((k2 < K) ? f0 : 0.0f);
        pr.y = (_Float16)((k2 + 1 < K) ? f1 : 0.0f);
        *(v2h*)(Bs + ((size_t)(k2 >> 5) * W + j) * CST + (k2 & 31)) = pr;
    }
    __syncthreads();

    if (m0 >= M) return;                     // wave-uniform (no later barriers)

    const int  arow  = m0 + lrow;
    const int  arowc = (arow < M) ? arow : (M - 1);
    const float* __restrict__ Arow = A + (size_t)arowc * K;

    // per-lane base of this lane's contiguous 16-half B run (tile 0, chunk 0)
    const _Float16* __restrict__ Bl = Bs + (size_t)lrow * CST + half * 16;

    const v8f vzero = {};
    v8f acc[NT];
#pragma unroll
    for (int t = 0; t < NT; ++t) acc[t] = vzero;

    int kc = 0;
    // ---- main: full 32-wide K chunks --------------------------------------
    for (; (kc + 1) * 32 <= K; ++kc) {
        const int k0 = kc * 32;
        v16h a;
        {
            const float4* p0 = (const float4*)(Arow + k0 + half * 16);
            const float4* p1 = (const float4*)(Arow + k0 + half * 16 + 8);
            float4 x0 = p0[0], x1 = p0[1];
            float4 y0 = p1[0], y1 = p1[1];
            a[0]  = (_Float16)x0.x; a[1]  = (_Float16)x0.y;
            a[2]  = (_Float16)x0.z; a[3]  = (_Float16)x0.w;
            a[4]  = (_Float16)x1.x; a[5]  = (_Float16)x1.y;
            a[6]  = (_Float16)x1.z; a[7]  = (_Float16)x1.w;
            a[8]  = (_Float16)y0.x; a[9]  = (_Float16)y0.y;
            a[10] = (_Float16)y0.z; a[11] = (_Float16)y0.w;
            a[12] = (_Float16)y1.x; a[13] = (_Float16)y1.y;
            a[14] = (_Float16)y1.z; a[15] = (_Float16)y1.w;
        }
        const _Float16* __restrict__ bk = Bl + (size_t)kc * W * CST;
#pragma unroll
        for (int t = 0; t < NT; ++t) {
            const _Float16* __restrict__ bp = bk + t * 16 * CST;
            const v8h b0 = *(const v8h*)bp;
            const v8h b1 = *(const v8h*)(bp + 8);
            const v16h b = __builtin_shufflevector(
                b0, b1, 0, 1, 2, 3, 4, 5, 6, 7, 8, 9, 10, 11, 12, 13, 14, 15);
            acc[t] = __builtin_amdgcn_wmma_f32_16x16x32_f16(
                false, a, false, b, (short)0, acc[t], false, false);
        }
    }
    // ---- K tail (A value-selects; B pad rows already zero in LDS) ---------
    if (kc * 32 < K) {
        const int k0 = kc * 32;
        v16h a;
#pragma unroll
        for (int e = 0; e < 16; ++e) {
            const int kk  = k0 + half * 16 + e;
            const float v = Arow[(kk < K) ? kk : (K - 1)];
            a[e] = (_Float16)((kk < K) ? v : 0.0f);
        }
        const _Float16* __restrict__ bk = Bl + (size_t)kc * W * CST;
#pragma unroll
        for (int t = 0; t < NT; ++t) {
            const _Float16* __restrict__ bp = bk + t * 16 * CST;
            const v8h b0 = *(const v8h*)bp;
            const v8h b1 = *(const v8h*)(bp + 8);
            const v16h b = __builtin_shufflevector(
                b0, b1, 0, 1, 2, 3, 4, 5, 6, 7, 8, 9, 10, 11, 12, 13, 14, 15);
            acc[t] = __builtin_amdgcn_wmma_f32_16x16x32_f16(
                false, a, false, b, (short)0, acc[t], false, false);
        }
    }
    // ---- epilogue: bias + activation + guarded stores ---------------------
#pragma unroll
    for (int t = 0; t < NT; ++t) {
        const int col = n0 + t * 16 + lrow;
        if (col >= N) continue;
        const float bv = bias ? bias[col] : 0.0f;
#pragma unroll
        for (int r = 0; r < 8; ++r) {
            const int row = m0 + r + 8 * half;
            float v = acc[t][r] + bv;
            if (act == ACT_CELU)      v = (v > 0.0f) ? v : (__expf(v) - 1.0f);
            else if (act == ACT_RELU) v = fmaxf(v, 0.0f);
            if (row < M) D[(size_t)row * N + col] = v;
        }
    }
}

// ---------------------------------------------------------------------------
// Small helper kernels
// ---------------------------------------------------------------------------
__global__ void k_fill(float* __restrict__ p, int n, float v)
{
    int i = blockIdx.x * blockDim.x + threadIdx.x;
    if (i < n) p[i] = v;
}

__global__ void k_deg_accum(float* __restrict__ deg,
                            const int* __restrict__ dst,
                            const float* __restrict__ w, int e_cnt)
{
    int e = blockIdx.x * blockDim.x + threadIdx.x;
    if (e < e_cnt) atomicAdd(&deg[dst[e]], w[e]);
}

__global__ void k_rsqrt_inplace(float* __restrict__ p, int n)
{
    int i = blockIdx.x * blockDim.x + threadIdx.x;
    if (i < n) p[i] = rsqrtf(p[i]);
}

// out[i,f] = dinv[i]^2 * t[i,f] + bias[f]   (self-loop + bias init)
__global__ void k_gcn_self(float* __restrict__ out,
                           const float* __restrict__ t,
                           const float* __restrict__ dinv,
                           const float* __restrict__ bias, int n, int F)
{
    int idx = blockIdx.x * blockDim.x + threadIdx.x;
    if (idx >= n * F) return;
    int i = idx / F, f = idx - i * F;
    float di = dinv[i];
    out[idx] = di * di * t[idx] + bias[f];
}

// out[dst,f] += dinv[src]*dinv[dst]*w * t[src,f]
__global__ void k_gcn_agg(float* __restrict__ out,
                          const float* __restrict__ t,
                          const float* __restrict__ dinv,
                          const int* __restrict__ src,
                          const int* __restrict__ dst,
                          const float* __restrict__ w, int e_cnt, int F)
{
    int idx = blockIdx.x * blockDim.x + threadIdx.x;
    if (idx >= e_cnt * F) return;
    int e = idx / F, f = idx - e * F;
    int s = src[e], d = dst[e];
    float c = dinv[s] * dinv[d] * w[e];
    atomicAdd(&out[(size_t)d * F + f], c * t[(size_t)s * F + f]);
}

__global__ void k_relu_inplace(float* __restrict__ p, int n)
{
    int i = blockIdx.x * blockDim.x + threadIdx.x;
    if (i < n) p[i] = fmaxf(p[i], 0.0f);
}

__global__ void k_clampmax_inplace(float* __restrict__ p, int n, float mx)
{
    int i = blockIdx.x * blockDim.x + threadIdx.x;
    if (i < n) p[i] = fminf(p[i], mx);
}

// z = mu + eps*exp(logstd); feat = [feat_x | z]; accumulate KL row sums.
__global__ void k_zfeat_kl(float* __restrict__ feat,
                           const float* __restrict__ featx,
                           const float* __restrict__ mu,
                           const float* __restrict__ ls,
                           const float* __restrict__ eps,
                           float* __restrict__ kl_accum, int n)
{
    int i = blockIdx.x * blockDim.x + threadIdx.x;
    float kl = 0.0f;
    if (i < n) {
#pragma unroll
        for (int f = 0; f < GAE1; ++f) {
            float m = mu[i * GAE1 + f];
            float l = ls[i * GAE1 + f];
            feat[i * FCAT + f]        = featx[i * DAE1 + f];
            feat[i * FCAT + DAE1 + f] = m + eps[i * GAE1 + f] * __expf(l);
            kl += 1.0f + 2.0f * l - m * m - __expf(2.0f * l);
        }
    }
    for (int off = 16; off > 0; off >>= 1) kl += __shfl_down(kl, off, 32);
    if ((threadIdx.x & 31) == 0 && kl != 0.0f) atomicAdd(kl_accum, kl);
}

// ct = rownorm(relu(feat @ Wc + bc))   one thread per node
__global__ void k_ct(float* __restrict__ ct,
                     const float* __restrict__ feat,
                     const float* __restrict__ Wc,
                     const float* __restrict__ bc, int n)
{
    int i = blockIdx.x * blockDim.x + threadIdx.x;
    if (i >= n) return;
    float fr[FCAT];
#pragma unroll
    for (int k = 0; k < FCAT; ++k) fr[k] = feat[(size_t)i * FCAT + k];
    float vals[CT], sum = 0.0f;
#pragma unroll
    for (int c = 0; c < CT; ++c) {
        float d = bc[c];
        for (int k = 0; k < FCAT; ++k) d += fr[k] * Wc[k * CT + c];
        d = fmaxf(d, 0.0f);
        vals[c] = d;
        sum += d;
    }
    float inv = 1.0f / (sum + NEPS);
#pragma unroll
    for (int c = 0; c < CT; ++c) ct[(size_t)i * CT + c] = vals[c] * inv;
}

// normalize each contiguous group of CT values of pw in place
__global__ void k_pw_norm(float* __restrict__ pw, int ngroups)
{
    int g = blockIdx.x * blockDim.x + threadIdx.x;
    if (g >= ngroups) return;
    float* p = pw + (size_t)g * CT;
    float sum = 0.0f, v[CT];
#pragma unroll
    for (int c = 0; c < CT; ++c) { v[c] = p[c]; sum += v[c]; }
    float inv = 1.0f / (sum + NEPS);
#pragma unroll
    for (int c = 0; c < CT; ++c) p[c] = v[c] * inv;
}

// sum over edges of (sigmoid(<zg[src],zg[dst]>) - w)^2  (w optional)
__global__ void k_edge_loss(const float* __restrict__ zg,
                            const int* __restrict__ src,
                            const int* __restrict__ dst,
                            const float* __restrict__ w,
                            int e_cnt, float* __restrict__ accum)
{
    int e = blockIdx.x * blockDim.x + threadIdx.x;
    float v = 0.0f;
    if (e < e_cnt) {
        const float* a = zg + (size_t)src[e] * FEAT;
        const float* b = zg + (size_t)dst[e] * FEAT;
        float d = 0.0f;
#pragma unroll 8
        for (int f = 0; f < FEAT; ++f) d += a[f] * b[f];
        float s = 1.0f / (1.0f + __expf(-d));
        float t = w ? (s - w[e]) : s;
        v = t * t;
    }
    for (int off = 16; off > 0; off >>= 1) v += __shfl_down(v, off, 32);
    if ((threadIdx.x & 31) == 0 && v != 0.0f) atomicAdd(accum, v);
}

// gae_loss = pos/E + neg/E + (-0.5*kl_sum/N)/N
__global__ void k_finalize(float* __restrict__ out, const float* __restrict__ sums)
{
    float pos = sums[0] / (float)EE;
    float neg = sums[1] / (float)EE;
    float kl  = -0.5f * sums[2] / (float)NN;
    out[0] = pos + neg + kl / (float)NN;
}

// ---------------------------------------------------------------------------
// Launcher
// ---------------------------------------------------------------------------
static inline dim3 gemm_grid(int M, int N, int NT)
{
    return dim3((N + 16 * NT - 1) / (16 * NT), ((M + 15) / 16 + 7) / 8);
}
static inline size_t gemm_lds(int K, int NT)
{
    const int KP = (K + 31) & ~31;
    return (size_t)(KP / 32) * 16 * NT * CST * sizeof(_Float16);
}

extern "C" void kernel_launch(void* const* d_in, const int* in_sizes, int n_in,
                              void* d_out, int out_size, void* d_ws, size_t ws_size,
                              hipStream_t stream)
{
    (void)in_sizes; (void)n_in; (void)out_size; (void)ws_size;

    const float* x    = (const float*)d_in[0];
    const float* ew   = (const float*)d_in[1];
    const float* eps  = (const float*)d_in[2];
    const float* W1   = (const float*)d_in[3];
    const float* b1   = (const float*)d_in[4];
    const float* W2   = (const float*)d_in[5];
    const float* b2   = (const float*)d_in[6];
    const float* Wf   = (const float*)d_in[7];
    const float* bf   = (const float*)d_in[8];
    const float* Wm   = (const float*)d_in[9];
    const float* bm   = (const float*)d_in[10];
    const float* Ws   = (const float*)d_in[11];
    const float* bs   = (const float*)d_in[12];
    const float* Wdec = (const float*)d_in[13];
    const float* bdec = (const float*)d_in[14];
    const float* Wg   = (const float*)d_in[15];
    const float* bg   = (const float*)d_in[16];
    const float* Wc   = (const float*)d_in[17];
    const float* bc   = (const float*)d_in[18];
    const float* Wv   = (const float*)d_in[19];
    const float* bv   = (const float*)d_in[20];
    const int*   ei   = (const int*)d_in[21];
    const int*   nei  = (const int*)d_in[22];
    const int* src  = ei;        const int* dst  = ei + EE;
    const int* nsrc = nei;       const int* ndst = nei + EE;

    float* out = (float*)d_out;

    // workspace layout (floats), ~13.2 MB total
    float* wsp   = (float*)d_ws;
    float* dinv  = wsp;                      // N
    float* h1    = dinv + NN;                // N*DAE0
    float* featx = h1 + (size_t)NN * DAE0;   // N*DAE1
    float* tbuf  = featx + (size_t)NN * DAE1;// N*GAE0 (reused scratch)
    float* h     = tbuf + (size_t)NN * GAE0; // N*GAE0
    float* mu    = h + (size_t)NN * GAE0;    // N*GAE1
    float* ls    = mu + (size_t)NN * GAE1;   // N*GAE1
    float* feat  = ls + (size_t)NN * GAE1;   // N*FCAT
    float* zg    = feat + (size_t)NN * FCAT; // N*FEAT
    float* sums  = zg + (size_t)NN * FEAT;   // 4: pos, neg, kl, spare

    const int T = 256;
    #define CDIV(a, b) (((a) + (b) - 1) / (b))

    // --- degree / normalization --------------------------------------------
    k_fill<<<CDIV(NN, T), T, 0, stream>>>(dinv, NN, 1.0f);           // deg = 1 (self loop)
    k_deg_accum<<<CDIV(EE, T), T, 0, stream>>>(dinv, dst, ew, EE);   // deg += sum w
    k_rsqrt_inplace<<<CDIV(NN, T), T, 0, stream>>>(dinv, NN);        // dinv = deg^-1/2

    // --- dense autoencoder: feat_x = celu(celu(x@W1+b1)@W2+b2) -------------
    k_gemm_wmma<2><<<gemm_grid(NN, DAE0, 2), T, gemm_lds(PP, 2), stream>>>(
        x, W1, b1, h1, NN, DAE0, PP, ACT_CELU);
    k_gemm_wmma<2><<<gemm_grid(NN, DAE1, 2), T, gemm_lds(DAE0, 2), stream>>>(
        h1, W2, b2, featx, NN, DAE1, DAE0, ACT_CELU);

    // --- GCN 1: h = relu(gcn(feat_x, Wf, bf)) ------------------------------
    k_gemm_wmma<2><<<gemm_grid(NN, GAE0, 2), T, gemm_lds(DAE1, 2), stream>>>(
        featx, Wf, nullptr, tbuf, NN, GAE0, DAE1, ACT_NONE);
    k_gcn_self<<<CDIV(NN * GAE0, T), T, 0, stream>>>(h, tbuf, dinv, bf, NN, GAE0);
    k_gcn_agg<<<CDIV(EE * GAE0, T), T, 0, stream>>>(h, tbuf, dinv, src, dst, ew, EE, GAE0);
    k_relu_inplace<<<CDIV(NN * GAE0, T), T, 0, stream>>>(h, NN * GAE0);

    // --- GCN 2: mu ---------------------------------------------------------
    k_gemm_wmma<2><<<gemm_grid(NN, GAE1, 2), T, gemm_lds(GAE0, 2), stream>>>(
        h, Wm, nullptr, tbuf, NN, GAE1, GAE0, ACT_NONE);
    k_gcn_self<<<CDIV(NN * GAE1, T), T, 0, stream>>>(mu, tbuf, dinv, bm, NN, GAE1);
    k_gcn_agg<<<CDIV(EE * GAE1, T), T, 0, stream>>>(mu, tbuf, dinv, src, dst, ew, EE, GAE1);

    // --- GCN 3: logstd = min(gcn, 10) --------------------------------------
    k_gemm_wmma<2><<<gemm_grid(NN, GAE1, 2), T, gemm_lds(GAE0, 2), stream>>>(
        h, Ws, nullptr, tbuf, NN, GAE1, GAE0, ACT_NONE);
    k_gcn_self<<<CDIV(NN * GAE1, T), T, 0, stream>>>(ls, tbuf, dinv, bs, NN, GAE1);
    k_gcn_agg<<<CDIV(EE * GAE1, T), T, 0, stream>>>(ls, tbuf, dinv, src, dst, ew, EE, GAE1);
    k_clampmax_inplace<<<CDIV(NN * GAE1, T), T, 0, stream>>>(ls, NN * GAE1, MAXLS);

    // --- reparameterize + concat + KL --------------------------------------
    k_fill<<<1, 4, 0, stream>>>(sums, 4, 0.0f);
    k_zfeat_kl<<<CDIV(NN, T), T, 0, stream>>>(feat, featx, mu, ls, eps, &sums[2], NN);

    // --- classifier head ----------------------------------------------------
    k_ct<<<CDIV(NN, T), T, 0, stream>>>(out + CT_OFF, feat, Wc, bc, NN);

    // --- decoder: x_dec = feat @ Wdec + bdec -------------------------------
    k_gemm_wmma<5><<<gemm_grid(NN, PP, 5), T, gemm_lds(FCAT, 5), stream>>>(
        feat, Wdec, bdec, out + XDEC_OFF, NN, PP, FCAT, ACT_NONE);

    // --- deconvolution head: pw (raw relu via WMMA, then group-normalize) --
    k_gemm_wmma<5><<<gemm_grid(NN, PP * CT, 5), T, gemm_lds(FCAT, 5), stream>>>(
        feat, Wv, bv, out + PW_OFF, NN, PP * CT, FCAT, ACT_RELU);
    k_pw_norm<<<CDIV(NN * PP, T), T, 0, stream>>>(out + PW_OFF, NN * PP);

    // --- graph recon losses -------------------------------------------------
    k_gemm_wmma<4><<<gemm_grid(NN, FEAT, 4), T, gemm_lds(FCAT, 4), stream>>>(
        feat, Wg, bg, zg, NN, FEAT, FCAT, ACT_RELU);
    k_edge_loss<<<CDIV(EE, T), T, 0, stream>>>(zg, src, dst, ew, EE, &sums[0]);
    k_edge_loss<<<CDIV(EE, T), T, 0, stream>>>(zg, nsrc, ndst, nullptr, EE, &sums[1]);
    k_finalize<<<1, 1, 0, stream>>>(out + GAE_OFF, sums);
}
